// MinLSTM_32581621907473
// MI455X (gfx1250) — matrix operand order repaired and checked by reading, not compiled, not run on hardware
//
#include <hip/hip_runtime.h>
#include <hip/hip_fp16.h>

typedef __attribute__((ext_vector_type(16))) _Float16 v16h;
typedef __attribute__((ext_vector_type(8)))  _Float16 v8h;
typedef __attribute__((ext_vector_type(4)))  _Float16 v4h;
typedef __attribute__((ext_vector_type(8)))  float    v8f;
typedef __attribute__((ext_vector_type(4)))  float    v4f;

#define NB 4
#define NS 8192
#define ND 1024
#define NH 1024
#define NBS (NB*NS)
#define NJ 2            /* h-subtiles per wave: N = 16*NJ = 32 columns */

__device__ __forceinline__ float softplusf(float x) {
  // stable: log(1+e^x) = log1p(e^{-|x|}) + max(x,0)
  return log1pf(__expf(-fabsf(x))) + fmaxf(x, 0.0f);
}

// ---------- f32 -> f16 conversion, 4 elements/thread ----------
__global__ void __launch_bounds__(256) k_cvt_f16(const float* __restrict__ src,
                                                 _Float16* __restrict__ dst, int n) {
  unsigned i = (blockIdx.x * 256u + threadIdx.x) * 4u;
  if (i + 3 < (unsigned)n) {
    v4f v = *(const v4f*)(src + i);
    *(v4h*)(dst + i) = __builtin_convertvector(v, v4h);
  }
}

__global__ void __launch_bounds__(256) k_zero(float* __restrict__ p, int n) {
  unsigned i = blockIdx.x * 256u + threadIdx.x;
  if (i < (unsigned)n) p[i] = 0.0f;
}

// ---------- fused  z = x·w_inᵀ  + gates + Σ_s reduction ----------
// Wave tile: 16 rows(s) x 32 h-cols x 3 gates (6 accumulators = 48 VGPRs,
// leaving ~48 VGPRs for in-flight load buffers). Block = 8 waves = 128 rows.
// grid = (NBS/128, NH/32)
template<bool XH, bool WH>
__global__ void __launch_bounds__(256) k_gates(
    const _Float16* __restrict__ xh, const float* __restrict__ xf,
    const _Float16* __restrict__ wh, const float* __restrict__ wf,
    float* __restrict__ acc, float* __restrict__ llf)
{
  const unsigned lane = threadIdx.x & 31u;
  const unsigned wave = threadIdx.x >> 5;
  const unsigned hi   = lane >> 4;      // which 16-lane half
  const unsigned ln   = lane & 15u;
  const unsigned rowBase = blockIdx.x * 128u + wave * 16u;   // global (b,s) row
  const unsigned hBase   = blockIdx.y * (16u * NJ);          // h column base

  const v8f vz = {0.f,0.f,0.f,0.f,0.f,0.f,0.f,0.f};
  v8f accv[NJ][3];
#pragma unroll
  for (int j = 0; j < NJ; ++j)
#pragma unroll
    for (int g = 0; g < 3; ++g) accv[j][g] = vz;

  // 32-bit half-element offsets -> zero-extended VGPR offset folded into
  // v_lshl_add_u64 with the uniform SGPR base (1 VALU per address).
  const unsigned xo = (rowBase + ln) * (unsigned)ND;
  const unsigned wobase = (hBase + ln) * (unsigned)ND;

  // Full unroll: 32 iters x (6 wmma + 14 b128 loads): scheduler batches loads
  // with partial s_wait_loadcnt and rotates B registers (no WAR-hazard nops).
#pragma unroll
  for (unsigned ks = 0; ks < ND / 32; ++ks) {
    // A fragment (16x32 f16): lane<16 holds K {k0..k0+15}, k0=32ks;
    // lane>=16 holds K {k0+16..k0+31}  (contiguous, matches B)
    const unsigned k0 = ks * 32u + hi * 16u;
    v8h a0, a1;
    if constexpr (XH) {
      a0 = *(const v8h*)(xh + (xo + k0));
      a1 = *(const v8h*)(xh + (xo + k0 + 8u));
    } else {
      a0 = __builtin_convertvector(*(const v8f*)(xf + (xo + k0)), v8h);
      a1 = __builtin_convertvector(*(const v8f*)(xf + (xo + k0 + 8u)), v8h);
    }
    v16h A = __builtin_shufflevector(a0, a1, 0,1,2,3,4,5,6,7,8,9,10,11,12,13,14,15);

    // B fragment (32x16 f16): lane = column (w_in row); K 0..15 in lanes<16, 16..31 above
    const unsigned kb = ks * 32u + hi * 16u;
#pragma unroll
    for (int j = 0; j < NJ; ++j) {
#pragma unroll
      for (int g = 0; g < 3; ++g) {
        const unsigned w0 = wobase + ((unsigned)g * NH + (unsigned)j * 16u) * (unsigned)ND + kb;
        v16h Bv;
        if constexpr (WH) {
          Bv = *(const v16h*)(wh + w0);
        } else {
          v8h b0 = __builtin_convertvector(*(const v8f*)(wf + w0), v8h);
          v8h b1 = __builtin_convertvector(*(const v8f*)(wf + (w0 + 8u)), v8h);
          Bv = __builtin_shufflevector(b0, b1, 0,1,2,3,4,5,6,7,8,9,10,11,12,13,14,15);
        }
        accv[j][g] = __builtin_amdgcn_wmma_f32_16x16x32_f16(
            false, A, false, Bv, (short)0, accv[j][g], false, false);
      }
    }
  }

  // Epilogue. C layout: element r of v8f in lane L -> row (r + 8*hi), col (L&15).
  // Algebra: log_i + log_g(h) - log_f = diff + log_g(h)   (sp(x)-sp(-x)=x),
  // with diff = sp(-f) - sp(-i);   log_f = -sp(diff) only needed at s==S-1.
  const unsigned b = rowBase >> 13;  // rowBase / NS (tile never crosses b)
#pragma unroll
  for (int j = 0; j < NJ; ++j) {
    float colsum = 0.0f;
#pragma unroll
    for (int r = 0; r < 8; ++r) {
      float fz = accv[j][0][r];
      float iz = accv[j][1][r];
      float hz = accv[j][2][r];
      float diff = softplusf(-fz) - softplusf(-iz);
      float lth = (hz >= 0.0f) ? __logf(hz + 0.5f) : -softplusf(-hz);  // log_g(h)
      colsum += __expf(diff + lth);
      unsigned rowG = rowBase + (unsigned)r + 8u * hi;
      if ((rowG & (NS - 1u)) == (NS - 1u))   // s == S-1: save log_f for final scale
        llf[b * NH + hBase + (unsigned)j * 16u + ln] = -softplusf(diff);
    }
    colsum += __shfl_xor(colsum, 16, 32);  // combine row-halves (same column)
    if (hi == 0)
      atomicAdd(&acc[b * NH + hBase + (unsigned)j * 16u + ln], colsum);
  }
}

// ---------- h_final = exp(log_f_last) * (0.5 + acc) ----------
__global__ void __launch_bounds__(256) k_hfinal(const float* __restrict__ acc,
                                                const float* __restrict__ llf,
                                                float* __restrict__ hf, int n) {
  unsigned i = blockIdx.x * 256u + threadIdx.x;
  if (i < (unsigned)n) hf[i] = __expf(llf[i]) * (0.5f + acc[i]);
}

// ---------- out[b,o] = h_final[b,:]·w_out[o,:] + b_out[o]  (4x1024, tiny) ----------
__global__ void __launch_bounds__(256) k_out(const float* __restrict__ hf,
                                             const float* __restrict__ wout,
                                             const float* __restrict__ bout,
                                             float* __restrict__ out) {
  unsigned idx = blockIdx.x * 256u + threadIdx.x;   // 0..4095
  unsigned b = idx >> 10, o = idx & 1023u;
  const v4f* hp = (const v4f*)(hf + b * NH);
  const v4f* wp = (const v4f*)(wout + (size_t)o * NH);
  float s = 0.0f;
#pragma unroll 4
  for (int k = 0; k < NH / 4; ++k) {
    v4f a = hp[k], w = wp[k];
    s += a.x * w.x + a.y * w.y + a.z * w.z + a.w * w.w;
  }
  out[idx] = s + bout[o];
}

extern "C" void kernel_launch(void* const* d_in, const int* in_sizes, int n_in,
                              void* d_out, int out_size, void* d_ws, size_t ws_size,
                              hipStream_t stream) {
  const float* x     = (const float*)d_in[0];   // [4,8192,1024]
  const float* w_in  = (const float*)d_in[1];   // [3072,1024]
  const float* w_out = (const float*)d_in[2];   // [1024,1024]
  const float* b_out = (const float*)d_in[3];   // [1024]
  float* out = (float*)d_out;                   // [4,1024]
  char* ws = (char*)d_ws;

  // workspace layout: [acc 16K][llf 16K][hf 16K][w_f16 6M][x_f16 64M]
  float* acc = (float*)(ws);
  float* llf = (float*)(ws + (size_t)NB * NH * 4);
  float* hf  = (float*)(ws + (size_t)2 * NB * NH * 4);
  const size_t off_wh = 49152;
  _Float16* wh = (_Float16*)(ws + off_wh);
  const size_t need_w = off_wh + (size_t)3 * NH * ND * 2;
  _Float16* xh = (_Float16*)(ws + need_w);
  const size_t need_x = need_w + (size_t)NBS * ND * 2;
  const bool useW = ws_size >= need_w;
  const bool useX = ws_size >= need_x;

  if (useW) k_cvt_f16<<<(3 * NH * ND / 4) / 256, 256, 0, stream>>>(w_in, wh, 3 * NH * ND);
  if (useX) k_cvt_f16<<<((size_t)NBS * ND / 4) / 256, 256, 0, stream>>>(x, xh, NBS * ND);
  k_zero<<<(NB * NH + 255) / 256, 256, 0, stream>>>(acc, NB * NH);

  dim3 grid(NBS / 128, NH / (16 * NJ));
  if (useX && useW)
    k_gates<true,  true ><<<grid, 256, 0, stream>>>(xh, x, wh, w_in, acc, llf);
  else if (useW)
    k_gates<false, true ><<<grid, 256, 0, stream>>>(xh, x, wh, w_in, acc, llf);
  else
    k_gates<false, false><<<grid, 256, 0, stream>>>(xh, x, wh, w_in, acc, llf);

  k_hfinal<<<(NB * NH) / 256, 256, 0, stream>>>(acc, llf, hf, NB * NH);
  k_out<<<(NB * 1024) / 256, 256, 0, stream>>>(hf, w_out, b_out, out);
}
